// GCNNetwork_75265006895959
// MI455X (gfx1250) — matrix operand order repaired and checked by reading, not compiled, not run on hardware
//
#include <hip/hip_runtime.h>

#define N_NODES 4096
#define IN_DIM  64
#define HID     256
#define EMB     128

typedef __attribute__((ext_vector_type(16))) _Float16 v16h;
typedef __attribute__((ext_vector_type(8)))  _Float16 h8;
typedef __attribute__((ext_vector_type(2)))  _Float16 vh2;
typedef __attribute__((ext_vector_type(8)))  float    v8f;
typedef __attribute__((ext_vector_type(2)))  float    vf2;

__device__ __forceinline__ float leakyf(float x) { return x >= 0.f ? x : 0.01f * x; }
__device__ __forceinline__ float reluf(float x)  { return x > 0.f ? x : 0.f; }

// ---------------------------------------------------------------------------
// Prep kernel: f32 -> f16 conversions / transposes so the WMMA kernel can use
// wide aligned vector loads for both A and B fragments.
//   featH[m][k]  = (f16) features[m][k]                  (row-major, 4096x64)
//   Wr1T[n][k]   = (f16) Wr1[k][n]                        (256x64)
//   Wr2T[n][k]   = (f16) Wr2[k][n]                        (128x256)
// ---------------------------------------------------------------------------
__global__ __launch_bounds__(256)
void prep_kernel(const float* __restrict__ feat,
                 const float* __restrict__ Wr1,
                 const float* __restrict__ Wr2,
                 _Float16* __restrict__ featH,
                 _Float16* __restrict__ Wr1T,
                 _Float16* __restrict__ Wr2T)
{
    const int tid    = blockIdx.x * blockDim.x + threadIdx.x;
    const int stride = gridDim.x * blockDim.x;
    for (int idx = tid; idx < N_NODES * IN_DIM; idx += stride)
        featH[idx] = (_Float16)feat[idx];
    for (int idx = tid; idx < HID * IN_DIM; idx += stride) {
        const int n = idx / IN_DIM, k = idx % IN_DIM;
        Wr1T[idx] = (_Float16)Wr1[(size_t)k * HID + n];
    }
    for (int idx = tid; idx < EMB * HID; idx += stride) {
        const int n = idx / HID, k = idx % HID;
        Wr2T[idx] = (_Float16)Wr2[(size_t)k * EMB + n];
    }
}

// ---------------------------------------------------------------------------
// Kernel A: raw = leaky(leaky(features @ Wr1 + br1) @ Wr2 + br2)
// One wave per block, 16 node rows per block, v_wmma_f32_16x16x32_f16.
// Fragment layouts used here (A and B share a contiguous k order):
//   A (16x32 f16):  lane L holds row m=L%16; elem v -> K = kk + 16*(L/16) + v
//                   => two contiguous 8-element runs (aligned b128 loads)
//   B (32x16 f16):  lane L holds col n=L%16; elem e -> K = kk + 16*(L/16) + e
//                   => one contiguous 16-element run from transposed weights
//   C/D (16x16 f32): lane L holds col n=L%16; VGPR r -> row m = r + 8*(L/16)
// ---------------------------------------------------------------------------
__global__ __launch_bounds__(32)
void raw_wmma_kernel(const _Float16* __restrict__ featH,
                     const _Float16* __restrict__ Wr1T, const float* __restrict__ br1,
                     const _Float16* __restrict__ Wr2T, const float* __restrict__ br2,
                     float* __restrict__ raw)
{
    __shared__ _Float16 h1[16 * HID];   // layer-1 activations, f16, 8 KB

    const int lane = threadIdx.x;
    const int m    = lane & 15;
    const int hi   = lane >> 4;
    const int row0 = blockIdx.x * 16;

    // ---- layer 1: [16 x 64] @ [64 x 256] + b, leaky ----
    for (int t = 0; t < HID / 16; ++t) {
        const int n = t * 16 + m;
        const float bias = br1[n];
        v8f acc = {bias, bias, bias, bias, bias, bias, bias, bias};
#pragma unroll
        for (int kk = 0; kk < IN_DIM; kk += 32) {
            const _Float16* ap = &featH[(size_t)(row0 + m) * IN_DIM + kk + 16 * hi];
            const h8 a0 = *(const h8*)ap;
            const h8 a1 = *(const h8*)(ap + 8);
            const v16h a = __builtin_shufflevector(a0, a1, 0, 1, 2, 3, 4, 5, 6, 7,
                                                   8, 9, 10, 11, 12, 13, 14, 15);
            const v16h b = *(const v16h*)&Wr1T[(size_t)n * IN_DIM + kk + 16 * hi];
            acc = __builtin_amdgcn_wmma_f32_16x16x32_f16(false, a, false, b,
                                                         (short)0, acc, false, false);
        }
#pragma unroll
        for (int r = 0; r < 8; ++r)
            h1[(r + 8 * hi) * HID + n] = (_Float16)leakyf(acc[r]);
    }
    __syncthreads();

    // ---- layer 2: [16 x 256] @ [256 x 128] + b, leaky ----
    for (int t = 0; t < EMB / 16; ++t) {
        const int n = t * 16 + m;
        const float bias = br2[n];
        v8f acc = {bias, bias, bias, bias, bias, bias, bias, bias};
#pragma unroll
        for (int kk = 0; kk < HID; kk += 32) {
            const _Float16* ap = &h1[m * HID + kk + 16 * hi];
            const h8 a0 = *(const h8*)ap;
            const h8 a1 = *(const h8*)(ap + 8);
            const v16h a = __builtin_shufflevector(a0, a1, 0, 1, 2, 3, 4, 5, 6, 7,
                                                   8, 9, 10, 11, 12, 13, 14, 15);
            const v16h b = *(const v16h*)&Wr2T[(size_t)n * HID + kk + 16 * hi];
            acc = __builtin_amdgcn_wmma_f32_16x16x32_f16(false, a, false, b,
                                                         (short)0, acc, false, false);
        }
#pragma unroll
        for (int r = 0; r < 8; ++r)
            raw[(size_t)(row0 + r + 8 * hi) * EMB + n] = leakyf(acc[r]);
    }
}

// ---------------------------------------------------------------------------
// Kernel B: the sequential scan. One 1024-thread workgroup (32 wave32s, one
// WGP). All four scan-phase weight matrices held f16 in LDS (256 KB of the
// 320 KB WGP LDS). Every mat-vec phase works on adjacent output PAIRS so each
// weight access is a packed 4-byte ds_load_b32 (full LDS bank width).
// descmask row is compacted (2% dense) before summing H rows; the next row is
// prefetched while this one is processed.
// ---------------------------------------------------------------------------
__global__ __launch_bounds__(1024, 1)
void scan_kernel(const int*   __restrict__ dm,
                 const int*   __restrict__ order,
                 const float* __restrict__ Wn1g, const float* __restrict__ bn1g,
                 const float* __restrict__ Wn2g, const float* __restrict__ bn2g,
                 const float* __restrict__ Ws1g, const float* __restrict__ bs1g,
                 const float* __restrict__ Ws2g, const float* __restrict__ bs2g,
                 const float* __restrict__ raw,
                 float*       __restrict__ H,
                 float*       __restrict__ out)
{
    extern __shared__ char smem[];
    _Float16* ws1 = (_Float16*)smem;          // 128x256
    _Float16* ws2 = ws1 + EMB * HID;          // 256x128
    _Float16* wn1 = ws2 + HID * EMB;          // 128x256
    _Float16* wn2 = wn1 + EMB * HID;          // 256x128
    float* bs1  = (float*)(wn2 + HID * EMB);  // 256
    float* bs2  = bs1 + HID;                  // 128
    float* bn1  = bs2 + EMB;                  // 256
    float* bn2  = bn1 + HID;                  // 128
    float* svec = bn2 + EMB;                  // 128
    float* act  = svec + EMB;                 // 256
    float* outv = act + HID;                  // 128
    float* red  = outv + EMB;                 // 2048
    int*   list = (int*)(red + 2048);         // 4096
    int*   cntp = list + N_NODES;             // 1 (+pad)

    const int tid = threadIdx.x;

    // -------- preload weights (f32 -> f16) and biases into LDS --------
    for (int idx = tid; idx < EMB * HID; idx += 1024) {
        ws1[idx] = (_Float16)Ws1g[idx];
        ws2[idx] = (_Float16)Ws2g[idx];
        wn1[idx] = (_Float16)Wn1g[idx];
        wn2[idx] = (_Float16)Wn2g[idx];
    }
    if (tid < HID) { bs1[tid] = bs1g[tid]; bn1[tid] = bn1g[tid]; }
    if (tid < EMB) { bs2[tid] = bs2g[tid]; bn2[tid] = bn2g[tid]; }
    if (tid == 0) *cntp = 0;
    __syncthreads();

    for (int step = 0; step < N_NODES; ++step) {
        const int i = order[step];            // N-1 .. 0 (reversed topo order)

        // -------- compact descmask row i (strictly upper: j > i only) -----
        for (int j = i + 1 + tid; j < N_NODES; j += 1024) {
            if (dm[(size_t)i * N_NODES + j] != 0) {
                const int p = atomicAdd(cntp, 1);
                list[p] = j;
            }
        }
        // prefetch next node's descmask row (pure hint; i-1 is next in order)
        if (i > 0) {
            const int off = tid * 16;         // 16 ints = 64 B per lane
            if (off < N_NODES - (i - 1))
                __builtin_prefetch(&dm[(size_t)(i - 1) * N_NODES + (i - 1) + off], 0, 3);
        }
        __syncthreads();
        const int cnt = *cntp;

        // -------- s = sum over active descendants of H[j]  (float2 lanes) --
        {
            const int cp = tid & 63;          // component pair 2cp, 2cp+1
            const int g  = tid >> 6;          // 0..15
            vf2 acc = {0.f, 0.f};
            for (int k = g; k < cnt; k += 16)
                acc += *(const vf2*)&H[(size_t)list[k] * EMB + 2 * cp];
            red[g * EMB + 2 * cp]     = acc[0];
            red[g * EMB + 2 * cp + 1] = acc[1];
        }
        __syncthreads();
        if (tid < EMB) {
            float p = 0.f;
#pragma unroll
            for (int g = 0; g < 16; ++g) p += red[g * EMB + tid];
            svec[tid] = p;
        }
        __syncthreads();

        // -------- summary layer 1: act = relu(s @ Ws1 + bs1) ---------------
        {
            const int op = tid & 127;         // output pair 2op, 2op+1
            const int g  = tid >> 7;          // 0..7
            const int k0 = g * 16;
            float a0 = 0.f, a1 = 0.f;
#pragma unroll
            for (int k = 0; k < 16; ++k) {
                const vh2 w = *(const vh2*)&ws1[(k0 + k) * HID + 2 * op];
                const float sv = svec[k0 + k];
                a0 += sv * (float)w[0];
                a1 += sv * (float)w[1];
            }
            red[g * HID + 2 * op]     = a0;
            red[g * HID + 2 * op + 1] = a1;
        }
        __syncthreads();
        if (tid < HID) {
            float p = bs1[tid];
#pragma unroll
            for (int g = 0; g < 8; ++g) p += red[g * HID + tid];
            act[tid] = reluf(p);
        }
        __syncthreads();

        // -------- summary layer 2, gate, out -------------------------------
        {
            const int cp = tid & 63;
            const int g  = tid >> 6;          // 0..15
            const int o0 = g * 16;
            float a0 = 0.f, a1 = 0.f;
#pragma unroll
            for (int o = 0; o < 16; ++o) {
                const vh2 w = *(const vh2*)&ws2[(o0 + o) * EMB + 2 * cp];
                const float av = act[o0 + o];
                a0 += av * (float)w[0];
                a1 += av * (float)w[1];
            }
            red[g * EMB + 2 * cp]     = a0;
            red[g * EMB + 2 * cp + 1] = a1;
        }
        __syncthreads();
        if (tid < EMB) {
            float p = bs2[tid];
#pragma unroll
            for (int g = 0; g < 16; ++g) p += red[g * EMB + tid];
            p = reluf(p);
            const float ov = raw[(size_t)i * EMB + tid] + (cnt > 0 ? p : 0.f);
            outv[tid] = ov;
            out[(size_t)i * EMB + tid] = ov;
        }
        __syncthreads();

        // -------- node layer 1: act = relu(out_i @ Wn1 + bn1) --------------
        {
            const int op = tid & 127;
            const int g  = tid >> 7;
            const int k0 = g * 16;
            float a0 = 0.f, a1 = 0.f;
#pragma unroll
            for (int k = 0; k < 16; ++k) {
                const vh2 w = *(const vh2*)&wn1[(k0 + k) * HID + 2 * op];
                const float ov = outv[k0 + k];
                a0 += ov * (float)w[0];
                a1 += ov * (float)w[1];
            }
            red[g * HID + 2 * op]     = a0;
            red[g * HID + 2 * op + 1] = a1;
        }
        __syncthreads();
        if (tid < HID) {
            float p = bn1[tid];
#pragma unroll
            for (int g = 0; g < 8; ++g) p += red[g * HID + tid];
            act[tid] = reluf(p);
        }
        __syncthreads();

        // -------- node layer 2 -> H[i] -------------------------------------
        {
            const int cp = tid & 63;
            const int g  = tid >> 6;
            const int o0 = g * 16;
            float a0 = 0.f, a1 = 0.f;
#pragma unroll
            for (int o = 0; o < 16; ++o) {
                const vh2 w = *(const vh2*)&wn2[(o0 + o) * EMB + 2 * cp];
                const float av = act[o0 + o];
                a0 += av * (float)w[0];
                a1 += av * (float)w[1];
            }
            red[g * EMB + 2 * cp]     = a0;
            red[g * EMB + 2 * cp + 1] = a1;
        }
        __syncthreads();
        if (tid < EMB) {
            float p = bn2[tid];
#pragma unroll
            for (int g = 0; g < 16; ++g) p += red[g * EMB + tid];
            H[(size_t)i * EMB + tid] = reluf(p);
        } else if (tid == 1023) {
            *cntp = 0;                         // reset counter for next node
        }
        __threadfence_block();   // make H[i] (global) visible to next iteration
        __syncthreads();
    }
}

// ---------------------------------------------------------------------------
extern "C" void kernel_launch(void* const* d_in, const int* in_sizes, int n_in,
                              void* d_out, int out_size, void* d_ws, size_t ws_size,
                              hipStream_t stream)
{
    (void)in_sizes; (void)n_in; (void)out_size; (void)ws_size;

    const float* feat  = (const float*)d_in[0];
    // d_in[1] = mask, unused by the reference
    const int*   dm    = (const int*)d_in[2];
    const int*   order = (const int*)d_in[3];
    const float* Wr1   = (const float*)d_in[4];
    const float* br1   = (const float*)d_in[5];
    const float* Wr2   = (const float*)d_in[6];
    const float* br2   = (const float*)d_in[7];
    const float* Wn1   = (const float*)d_in[8];
    const float* bn1   = (const float*)d_in[9];
    const float* Wn2   = (const float*)d_in[10];
    const float* bn2   = (const float*)d_in[11];
    const float* Ws1   = (const float*)d_in[12];
    const float* bs1   = (const float*)d_in[13];
    const float* Ws2   = (const float*)d_in[14];
    const float* bs2   = (const float*)d_in[15];

    float*    raw   = (float*)d_ws;                         // 4096*128 f32 (2 MB)
    float*    H     = raw + (size_t)N_NODES * EMB;          // 4096*128 f32 (2 MB)
    _Float16* featH = (_Float16*)(H + (size_t)N_NODES * EMB); // 4096*64 f16
    _Float16* Wr1T  = featH + (size_t)N_NODES * IN_DIM;       // 256*64  f16
    _Float16* Wr2T  = Wr1T + (size_t)HID * IN_DIM;            // 128*256 f16

    float* out = (float*)d_out;

    // Phase 0: f16 convert + transpose so WMMA fragments are wide aligned loads
    prep_kernel<<<512, 256, 0, stream>>>(feat, Wr1, Wr2, featH, Wr1T, Wr2T);

    // Phase 1: fully parallel raw_network via WMMA (256 blocks x 1 wave)
    raw_wmma_kernel<<<N_NODES / 16, 32, 0, stream>>>(featH, Wr1T, br1, Wr2T, br2, raw);

    // Phase 2: sequential scan on one WGP with LDS-resident f16 weights
    const size_t shmem =
        (size_t)4 * EMB * HID * sizeof(_Float16)                    // weights (256 KB)
      + (size_t)(HID + EMB + HID + EMB + EMB + HID + EMB + 2048) * sizeof(float)
      + (size_t)(N_NODES + 4) * sizeof(int);
    scan_kernel<<<1, 1024, shmem, stream>>>(dm, order,
                                            Wn1, bn1, Wn2, bn2,
                                            Ws1, bs1, Ws2, bs2,
                                            raw, H, out);
}
